// MHA_90529320665271
// MI455X (gfx1250) — matrix operand order repaired and checked by reading, not compiled, not run on hardware
//
#include <hip/hip_runtime.h>

// ---------------------------------------------------------------------------
// MHA (B=2, S=2048, D=2048, H=16, Dh=128) for gfx1250 (MI455X, wave32, WMMA)
//   0) one-time f32 -> f16 casts of q/k/v and the 4 weight matrices
//   1) Q/K projections -> f16 [B,H,S,Dh]; V projection -> f16 [B,H,Dh,S]
//   2) fused flash attention; tile staging via Tensor Data Mover (TDM)
//      descriptors (fallback: async-to-LDS), K/V double-buffered
//   3) output projection: f16 attn @ Wo^T -> f32 d_out
// ---------------------------------------------------------------------------

typedef __attribute__((ext_vector_type(16))) _Float16     v16h;
typedef __attribute__((ext_vector_type(8)))  float        v8f;
typedef __attribute__((ext_vector_type(4)))  int          v4i;
typedef __attribute__((ext_vector_type(8)))  int          v8i;
typedef __attribute__((ext_vector_type(4)))  unsigned int v4u;

union Frag {
  v16h v;
  uint4 q[2];
};

__device__ __forceinline__ unsigned short f32_f16(float f) {
  _Float16 h = (_Float16)f;
  return __builtin_bit_cast(unsigned short, h);
}

__device__ __forceinline__ float hmax16(float v) {
#pragma unroll
  for (int m = 1; m < 16; m <<= 1) v = fmaxf(v, __shfl_xor(v, m, 32));
  return v;
}
__device__ __forceinline__ float hsum16(float v) {
#pragma unroll
  for (int m = 1; m < 16; m <<= 1) v += __shfl_xor(v, m, 32);
  return v;
}

__device__ __forceinline__ v8f wmma16(const Frag& a, const Frag& b, v8f c) {
  return __builtin_amdgcn_wmma_f32_16x16x32_f16(
      false, a.v, false, b.v, (short)0, c, false, false);
}

// ---- CDNA5 async global -> LDS copy (ASYNCcnt path, probe-verified) -------
__device__ __forceinline__ void async_cp16(unsigned short* lds,
                                           const unsigned short* g) {
#if __has_builtin(__builtin_amdgcn_global_load_async_to_lds_b128)
  typedef __attribute__((address_space(1))) v4i GVi;
  typedef __attribute__((address_space(3))) v4i LVi;
  __builtin_amdgcn_global_load_async_to_lds_b128((GVi*)g, (LVi*)lds, 0, 0);
#else
  asm volatile("global_load_async_to_lds_b128 %0, %1, off"
               :: "v"((unsigned)(unsigned long long)lds), "v"(g)
               : "memory");
#endif
}

__device__ __forceinline__ void wait_async0() {
#if __has_builtin(__builtin_amdgcn_s_wait_asynccnt)
  __builtin_amdgcn_s_wait_asynccnt(0);
#else
  asm volatile("s_wait_asynccnt 0x0" ::: "memory");
#endif
}

// ---- CDNA5 Tensor Data Mover: 2D tile load with LDS padding ---------------
#if __has_builtin(__builtin_amdgcn_tensor_load_to_lds)
#define HAVE_TDM 1
#else
#define HAVE_TDM 0
#endif

__device__ __forceinline__ void wait_tensor0() {
#if __has_builtin(__builtin_amdgcn_s_wait_tensorcnt)
  __builtin_amdgcn_s_wait_tensorcnt(0);
#else
  asm volatile("s_wait_tensorcnt 0x0" ::: "memory");
#endif
}

#if HAVE_TDM
// D# packing per CDNA5 ISA ch.8.3/8.4. 2D tensor/tile, data_size = 2 bytes,
// LDS padding enabled (padi: 3->16DW, 4->32DW, 5->64DW; pada: 3 -> 4 DWORDs).
__device__ __forceinline__ void tdm_load_2d(unsigned lds_addr, const void* gtile,
                                            unsigned td0, unsigned td1,
                                            unsigned tile0, unsigned tile1,
                                            unsigned long long stride0,
                                            unsigned padi, unsigned pada) {
  const unsigned long long ga = (unsigned long long)gtile;
  v4u g0;
  g0[0] = 1u;                                             // count=1, user D#
  g0[1] = lds_addr;                                       // LDS byte address
  g0[2] = (unsigned)ga;                                   // global_addr[31:0]
  g0[3] = (unsigned)((ga >> 32) & 0x01FFFFFFu) | (2u << 30);  // [56:32] | type=2
  v8i g1;
  g1[0] = (int)((1u << 16) | (1u << 20) | (padi << 22) | (pada << 25));
  g1[1] = (int)((td0 & 0xFFFFu) << 16);                   // tensor_dim0 lo
  g1[2] = (int)(((td0 >> 16) & 0xFFFFu) | ((td1 & 0xFFFFu) << 16));
  g1[3] = (int)(((td1 >> 16) & 0xFFFFu) | (tile0 << 16)); // tile_dim0
  g1[4] = (int)(tile1 & 0xFFFFu);                         // tile_dim1
  g1[5] = (int)(unsigned)(stride0 & 0xFFFFFFFFull);       // dim0 stride lo
  g1[6] = (int)(unsigned)((stride0 >> 32) & 0xFFFFull);   // dim0 stride hi
  g1[7] = 0;
  v4i z4 = {0, 0, 0, 0};
#if __clang_major__ >= 23
  v8i z8 = {0, 0, 0, 0, 0, 0, 0, 0};
  __builtin_amdgcn_tensor_load_to_lds(g0, g1, z4, z4, z8, 0);
#else
  __builtin_amdgcn_tensor_load_to_lds(g0, g1, z4, z4, 0);
#endif
}
#endif

__device__ __forceinline__ unsigned lds_addr_of(const void* p) {
  return (unsigned)(unsigned long long)p;
}

// ---------------------------------------------------------------------------
// cast: f32 -> f16, vectorized x4 (n4 = n/4)
// ---------------------------------------------------------------------------
__global__ __launch_bounds__(256)
void cast_f32_f16_kernel(const float* __restrict__ src,
                         unsigned short* __restrict__ dst, int n4) {
  int i = blockIdx.x * blockDim.x + threadIdx.x;
  const int stride = gridDim.x * blockDim.x;
  for (; i < n4; i += stride) {
    const float4 f = ((const float4*)src)[i];
    alignas(8) unsigned short h[4] = {f32_f16(f.x), f32_f16(f.y),
                                      f32_f16(f.z), f32_f16(f.w)};
    ((unsigned long long*)dst)[i] = *(const unsigned long long*)h;
  }
}

// ---------------------------------------------------------------------------
// GEMM: Y[M,N] = X[M,K](f16) * W[N,K](f16)^T
// MODE 0: f16 [B,H,S,Dh] | MODE 1: f16 [B,H,Dh,S] | MODE 2: f32 [M,N]
// Block 256 thr (8 waves); macro tile 128x128; K-step 32; wave tile 32x64.
// Double-buffered LDS; staging via TDM (wave 0 issues 2 descriptors/step).
// ---------------------------------------------------------------------------
template<int MODE>
__global__ __launch_bounds__(256)
void gemm_xwt(const unsigned short* __restrict__ X,
              const unsigned short* __restrict__ W,
              void* __restrict__ Y, int M, int N, int K)
{
  __shared__ __align__(16) unsigned short Xs[2][128 * 40];  // 80B row stride
  __shared__ __align__(16) unsigned short Ws[2][128 * 40];

  const int t    = threadIdx.x;
  const int lane = t & 31;
  const int wave = t >> 5;
  const int m0   = blockIdx.x * 128;
  const int n0   = blockIdx.y * 128;
  const int wm   = (wave & 3) * 32;
  const int wn   = (wave >> 2) * 64;
  const int fm   = lane & 15;
  const int fh   = lane >> 4;

  v8f acc[2][4] = {};

#if HAVE_TDM
  auto stage = [&](int buf, int k0) {  // call from wave 0 only
    tdm_load_2d(lds_addr_of(&Xs[buf][0]), X + (size_t)m0 * K + k0,
                (unsigned)K, (unsigned)M, 32, 128, (unsigned long long)K, 3, 3);
    tdm_load_2d(lds_addr_of(&Ws[buf][0]), W + (size_t)n0 * K + k0,
                (unsigned)K, (unsigned)N, 32, 128, (unsigned long long)K, 3, 3);
  };
  if (wave == 0) stage(0, 0);
#else
  const int xr = t >> 1, xh = t & 1;
  auto stage = [&](int buf, int k0) {
    const unsigned short* gx = X + (size_t)(m0 + xr) * K + k0 + xh * 16;
    async_cp16(&Xs[buf][xr * 40 + xh * 16], gx);
    async_cp16(&Xs[buf][xr * 40 + xh * 16 + 8], gx + 8);
    const unsigned short* gw = W + (size_t)(n0 + xr) * K + k0 + xh * 16;
    async_cp16(&Ws[buf][xr * 40 + xh * 16], gw);
    async_cp16(&Ws[buf][xr * 40 + xh * 16 + 8], gw + 8);
  };
  stage(0, 0);
#endif

  for (int k0 = 0; k0 < K; k0 += 32) {
    const int buf = (k0 >> 5) & 1;
#if HAVE_TDM
    if (wave == 0) wait_tensor0();
#else
    wait_async0();
#endif
    __syncthreads();
    if (k0 + 32 < K) {
#if HAVE_TDM
      if (wave == 0) stage(buf ^ 1, k0 + 32);
#else
      stage(buf ^ 1, k0 + 32);
#endif
    }

    const unsigned short* xs = Xs[buf];
    const unsigned short* ws = Ws[buf];
    Frag a[2], b[4];
#pragma unroll
    for (int i = 0; i < 2; ++i) {
      const int row = wm + i * 16 + fm;
      a[i].q[0] = *(const uint4*)&xs[row * 40 + fh * 8];
      a[i].q[1] = *(const uint4*)&xs[row * 40 + 16 + fh * 8];
    }
#pragma unroll
    for (int j = 0; j < 4; ++j) {
      const int wrow = wn + j * 16 + fm;
      b[j].q[0] = *(const uint4*)&ws[wrow * 40 + fh * 8];
      b[j].q[1] = *(const uint4*)&ws[wrow * 40 + 16 + fh * 8];
    }
#pragma unroll
    for (int i = 0; i < 2; ++i)
#pragma unroll
      for (int j = 0; j < 4; ++j)
        acc[i][j] = wmma16(a[i], b[j], acc[i][j]);
  }

  // store (C layout: vgpr r -> M=r+8*fh, lane -> N=fm)
#pragma unroll
  for (int i = 0; i < 2; ++i) {
#pragma unroll
    for (int j = 0; j < 4; ++j) {
#pragma unroll
      for (int r = 0; r < 8; ++r) {
        const int gm = m0 + wm + i * 16 + r + 8 * fh;
        const int gn = n0 + wn + j * 16 + fm;
        const float val = acc[i][j][r];
        const int bb = gm >> 11, ss = gm & 2047;  // S = 2048
        const int hh = gn >> 7,  dd = gn & 127;   // Dh = 128
        if constexpr (MODE == 0) {
          ((unsigned short*)Y)[((((size_t)bb * 16 + hh) * 2048) + ss) * 128 + dd] =
              f32_f16(val);
        } else if constexpr (MODE == 1) {
          ((unsigned short*)Y)[((((size_t)bb * 16 + hh) * 128) + dd) * 2048 + ss] =
              f32_f16(val);
        } else {
          ((float*)Y)[(size_t)gm * N + gn] = val;
        }
      }
    }
  }
}

// ---------------------------------------------------------------------------
// Fused flash attention. grid = (S/64 q-tiles, B*H). block = 256 (8 waves).
// ---------------------------------------------------------------------------
__global__ __launch_bounds__(256)
void flash_attn_kernel(const unsigned short* __restrict__ Qh,   // [B,H,S,Dh]
                       const unsigned short* __restrict__ Kh,   // [B,H,S,Dh]
                       const unsigned short* __restrict__ VhT,  // [B,H,Dh,S]
                       const unsigned char* __restrict__ mask,
                       unsigned short* __restrict__ out,        // [B,S,D] f16
                       float scale)
{
  constexpr int S = 2048, Dh = 128;
  __shared__ __align__(16) unsigned short Qs [64 * 136];       // 272B stride
  __shared__ __align__(16) unsigned short Ks [2][64 * 136];
  __shared__ __align__(16) unsigned short Vts[2][128 * 72];    // 144B stride
  __shared__ __align__(16) unsigned short Ps [64 * 72];
  __shared__ float red_max[8][16];
  __shared__ float red_sum[8][16];

  const int t    = threadIdx.x;
  const int lane = t & 31;
  const int wave = t >> 5;
  const int fm   = lane & 15;
  const int fh   = lane >> 4;
  const int bh   = blockIdx.y;
  const int b    = bh >> 4;
  const int h    = bh & 15;
  const int q0   = blockIdx.x * 64;
  const size_t headoff = (size_t)bh * S * Dh;  // same for [S,Dh] and [Dh,S]

  const int wm = (wave & 3) * 16;   // q-row tile base
  const int wk = (wave >> 2) * 32;  // score key-col base (0/32)
  const int wn = (wave >> 2) * 64;  // O dh-col base (0/64)

#if HAVE_TDM
  if (wave == 0) {
    tdm_load_2d(lds_addr_of(&Qs[0]), Qh + headoff + (size_t)q0 * Dh,
                Dh, S, 128, 64, Dh, 5, 3);
  }
  auto stageKV = [&](int buf, int kb) {  // call from wave 0 only
    tdm_load_2d(lds_addr_of(&Ks[buf][0]), Kh + headoff + (size_t)kb * Dh,
                Dh, S, 128, 64, Dh, 5, 3);
    tdm_load_2d(lds_addr_of(&Vts[buf][0]), VhT + headoff + kb,
                S, Dh, 64, 128, S, 4, 3);
  };
  if (wave == 0) stageKV(0, 0);
#else
  const int sr  = t >> 2;  // K staging row 0..63
  const int seg = t & 3;
  const int vr  = t >> 1;  // V^T staging row 0..127
  const int vhf = t & 1;
  {
    const unsigned short* g = Qh + headoff + (size_t)(q0 + sr) * Dh + seg * 32;
    unsigned short* d = &Qs[sr * 136 + seg * 32];
#pragma unroll
    for (int j = 0; j < 4; ++j) async_cp16(d + j * 8, g + j * 8);
  }
  auto stageKV = [&](int buf, int kb) {
    const unsigned short* gk = Kh + headoff + (size_t)(kb + sr) * Dh + seg * 32;
    unsigned short* dk = &Ks[buf][sr * 136 + seg * 32];
#pragma unroll
    for (int j = 0; j < 4; ++j) async_cp16(dk + j * 8, gk + j * 8);
    const unsigned short* gv = VhT + headoff + (size_t)vr * S + kb + vhf * 32;
    unsigned short* dv = &Vts[buf][vr * 72 + vhf * 32];
#pragma unroll
    for (int j = 0; j < 4; ++j) async_cp16(dv + j * 8, gv + j * 8);
  };
  stageKV(0, 0);
#endif

  v8f o_acc[4] = {};
  float m_i[8], l_i[8];
#pragma unroll
  for (int r = 0; r < 8; ++r) { m_i[r] = -__builtin_inff(); l_i[r] = 0.f; }

  for (int kb = 0; kb < S; kb += 64) {
    const int buf = (kb >> 6) & 1;
#if HAVE_TDM
    if (wave == 0) wait_tensor0();
#else
    wait_async0();
#endif
    __syncthreads();
    if (kb + 64 < S) {
#if HAVE_TDM
      if (wave == 0) stageKV(buf ^ 1, kb + 64);
#else
      stageKV(buf ^ 1, kb + 64);
#endif
    }

    const unsigned short* ks = Ks[buf];
    const unsigned short* vs = Vts[buf];

    // ---- scores: 2 tiles of 16x16 per wave, K-dim = Dh = 128 ----
    v8f sc[2] = {};
#pragma unroll
    for (int kk = 0; kk < 4; ++kk) {
      Frag aq;
      const int qrow = wm + fm;
      aq.q[0] = *(const uint4*)&Qs[qrow * 136 + kk * 32 + fh * 8];
      aq.q[1] = *(const uint4*)&Qs[qrow * 136 + kk * 32 + 16 + fh * 8];
#pragma unroll
      for (int nt = 0; nt < 2; ++nt) {
        Frag bk;
        const int krow = wk + nt * 16 + fm;
        bk.q[0] = *(const uint4*)&ks[krow * 136 + kk * 32 + fh * 8];
        bk.q[1] = *(const uint4*)&ks[krow * 136 + kk * 32 + 16 + fh * 8];
        sc[nt] = wmma16(aq, bk, sc[nt]);
      }
    }

    // ---- scale + mask + per-wave row max over its 32 cols ----
    float mrow[8];
#pragma unroll
    for (int r = 0; r < 8; ++r) {
      const int qg = q0 + wm + r + 8 * fh;
      float s0 = sc[0][r] * scale;
      float s1 = sc[1][r] * scale;
      const int kg0 = kb + wk + fm;
      if (mask[((size_t)b * S + qg) * S + kg0])      s0 = -__builtin_inff();
      if (mask[((size_t)b * S + qg) * S + kg0 + 16]) s1 = -__builtin_inff();
      sc[0][r] = s0; sc[1][r] = s1;
      mrow[r] = hmax16(fmaxf(s0, s1));
    }
    if (fm == 0) {
#pragma unroll
      for (int r = 0; r < 8; ++r) red_max[wave][r + 8 * fh] = mrow[r];
    }
    __syncthreads();

    // ---- online softmax update ----
    float alpha[8];
#pragma unroll
    for (int r = 0; r < 8; ++r) {
      const int row = r + 8 * fh;
      const float mc = fmaxf(red_max[wave & 3][row], red_max[(wave & 3) + 4][row]);
      const float mn = fmaxf(m_i[r], mc);
      alpha[r] = (m_i[r] == -__builtin_inff()) ? 0.f : __expf(m_i[r] - mn);
      m_i[r] = mn;
      l_i[r] *= alpha[r];
    }

    // ---- P = exp(s - m) -> LDS (f16) + partial row sums ----
    float srow[8];
#pragma unroll
    for (int r = 0; r < 8; ++r) {
      const float p0 = __expf(sc[0][r] - m_i[r]);
      const float p1 = __expf(sc[1][r] - m_i[r]);
      const int prow = wm + r + 8 * fh;
      Ps[prow * 72 + wk + fm]      = f32_f16(p0);
      Ps[prow * 72 + wk + 16 + fm] = f32_f16(p1);
      srow[r] = hsum16(p0 + p1);
    }
    if (fm == 0) {
#pragma unroll
      for (int r = 0; r < 8; ++r) red_sum[wave][r + 8 * fh] = srow[r];
    }
    __syncthreads();
#pragma unroll
    for (int r = 0; r < 8; ++r) {
      const int row = r + 8 * fh;
      l_i[r] += red_sum[wave & 3][row] + red_sum[(wave & 3) + 4][row];
    }

    // ---- rescale O, then O += P * V ----
#pragma unroll
    for (int j = 0; j < 4; ++j)
#pragma unroll
      for (int r = 0; r < 8; ++r) o_acc[j][r] *= alpha[r];

#pragma unroll
    for (int kk = 0; kk < 2; ++kk) {
      Frag ap;
      const int prow = wm + fm;
      ap.q[0] = *(const uint4*)&Ps[prow * 72 + kk * 32 + fh * 8];
      ap.q[1] = *(const uint4*)&Ps[prow * 72 + kk * 32 + 16 + fh * 8];
#pragma unroll
      for (int j = 0; j < 4; ++j) {
        Frag bv;
        const int vrow = wn + j * 16 + fm;
        bv.q[0] = *(const uint4*)&vs[vrow * 72 + kk * 32 + fh * 8];
        bv.q[1] = *(const uint4*)&vs[vrow * 72 + kk * 32 + 16 + fh * 8];
        o_acc[j] = wmma16(ap, bv, o_acc[j]);
      }
    }
    // no trailing barrier: next iteration's top barrier orders P reuse,
    // and K/V tiles are double-buffered.
  }

  // ---- epilogue ----
#pragma unroll
  for (int j = 0; j < 4; ++j) {
#pragma unroll
    for (int r = 0; r < 8; ++r) {
      const int qg = q0 + wm + r + 8 * fh;
      const int dh = wn + j * 16 + fm;
      const float val = o_acc[j][r] / l_i[r];
      out[((size_t)b * S + qg) * 2048 + h * Dh + dh] = f32_f16(val);
    }
  }
}

// ---------------------------------------------------------------------------
extern "C" void kernel_launch(void* const* d_in, const int* in_sizes, int n_in,
                              void* d_out, int out_size, void* d_ws, size_t ws_size,
                              hipStream_t stream) {
  (void)in_sizes; (void)n_in; (void)out_size; (void)ws_size;
  constexpr int S = 2048, D = 2048;
  const float* q  = (const float*)d_in[0];
  const float* k  = (const float*)d_in[1];
  const float* v  = (const float*)d_in[2];
  const unsigned char* mask = (const unsigned char*)d_in[3];
  const float* Wq = (const float*)d_in[4];
  const float* Wk = (const float*)d_in[5];
  const float* Wv = (const float*)d_in[6];
  const float* Wo = (const float*)d_in[7];

  const size_t NX = (size_t)2 * S * D;
  const size_t NW = (size_t)D * D;

  char* ws = (char*)d_ws;
  size_t off = 0;
  auto take = [&](size_t elems) {
    unsigned short* p = (unsigned short*)(ws + off); off += elems * 2; return p;
  };
  unsigned short* qh16 = take(NX);
  unsigned short* kh16 = take(NX);
  unsigned short* vh16 = take(NX);
  unsigned short* Wq16 = take(NW);
  unsigned short* Wk16 = take(NW);
  unsigned short* Wv16 = take(NW);
  unsigned short* Wo16 = take(NW);
  unsigned short* Qh   = take(NX);   // f16 [B,H,S,Dh]
  unsigned short* Kh   = take(NX);   // f16 [B,H,S,Dh]
  unsigned short* VhT  = take(NX);   // f16 [B,H,Dh,S]
  unsigned short* At   = take(NX);   // f16 [B,S,D]

  dim3 blk(256);
  const int cb = 1024;
  cast_f32_f16_kernel<<<cb, blk, 0, stream>>>(q,  qh16, (int)(NX / 4));
  cast_f32_f16_kernel<<<cb, blk, 0, stream>>>(k,  kh16, (int)(NX / 4));
  cast_f32_f16_kernel<<<cb, blk, 0, stream>>>(v,  vh16, (int)(NX / 4));
  cast_f32_f16_kernel<<<cb, blk, 0, stream>>>(Wq, Wq16, (int)(NW / 4));
  cast_f32_f16_kernel<<<cb, blk, 0, stream>>>(Wk, Wk16, (int)(NW / 4));
  cast_f32_f16_kernel<<<cb, blk, 0, stream>>>(Wv, Wv16, (int)(NW / 4));
  cast_f32_f16_kernel<<<cb, blk, 0, stream>>>(Wo, Wo16, (int)(NW / 4));

  dim3 gproj(32, 16);  // M/128 = 32, N/128 = 16
  gemm_xwt<0><<<gproj, blk, 0, stream>>>(qh16, Wq16, Qh,  4096, 2048, 2048);
  gemm_xwt<0><<<gproj, blk, 0, stream>>>(kh16, Wk16, Kh,  4096, 2048, 2048);
  gemm_xwt<1><<<gproj, blk, 0, stream>>>(vh16, Wv16, VhT, 4096, 2048, 2048);

  const float scale = 0.08838834764831845f;  // 1/sqrt(128)
  flash_attn_kernel<<<dim3(32, 32), blk, 0, stream>>>(Qh, Kh, VhT, mask, At, scale);

  gemm_xwt<2><<<gproj, blk, 0, stream>>>(At, Wo16, d_out, 4096, 2048, 2048);
}
